// CFNetInteractionBlock_38826504356690
// MI455X (gfx1250) — matrix operand order repaired and checked by reading, not compiled, not run on hardware
//
#include <hip/hip_runtime.h>
#include <stdint.h>

#define F   128
#define LDA 136   // padded LDS row stride (bf16 elements) -> conflict-free frag loads

typedef __attribute__((ext_vector_type(16))) __bf16 v16bf;
typedef __attribute__((ext_vector_type(8)))  __bf16 v8bf;
typedef __attribute__((ext_vector_type(8)))  float  v8f;

__device__ __forceinline__ unsigned short f2bf(float f) {
  unsigned int u = __float_as_uint(f);
  u += 0x7FFFu + ((u >> 16) & 1u);          // round-to-nearest-even
  return (unsigned short)(u >> 16);
}

__device__ __forceinline__ float ssp(float x) {
  // softplus(x) - ln2, numerically stable
  float t = __expf(-fabsf(x));
  return fmaxf(x, 0.0f) + __logf(1.0f + t) - 0.69314718055994531f;
}

// 16x32 bf16 A-fragment from LDS tile (row-major, stride LDA).
// Contiguous k order (same as B): lanes 0-15 -> M=lane, K=kbase..kbase+15; lanes 16-31 -> K+16.
__device__ __forceinline__ v16bf load_afrag(const unsigned short* s, int m, int kbase, int lane) {
  int koff = (lane >> 4) << 4;
  const v8bf* lo = (const v8bf*)(s + m * LDA + kbase + koff);
  const v8bf* hi = (const v8bf*)(s + m * LDA + kbase + koff + 8);
  v8bf a = *lo, b = *hi;
  return __builtin_shufflevector(a, b, 0,1,2,3,4,5,6,7,8,9,10,11,12,13,14,15);
}

// 32x16 bf16 B-fragment from pre-transposed weights Wt[n][k] (global, stride F).
// Contiguous k order (same as A): lanes 0-15 -> N=lane, K=kbase..kbase+15; lanes 16-31 -> K+16.
__device__ __forceinline__ v16bf load_bfrag(const unsigned short* wt, int n, int kbase, int lane) {
  int ks = kbase + ((lane >> 4) << 4);
  return *(const v16bf*)(wt + n * F + ks);
}

// Stage a 128x128 fp32 tile -> bf16 LDS tile (zero-fill rows >= limit).
__device__ __forceinline__ void stage_tile(unsigned short* sA, const float* src,
                                           int rowBase, int limit, int tid) {
  if (rowBase + 128 <= limit) {                      // full tile: no per-row guards
    for (int i = tid * 4; i < 128 * 128; i += 1024) {
      int r = i >> 7, c = i & 127;
      float4 v = *(const float4*)(src + (size_t)(rowBase + r) * F + c);
      sA[r * LDA + c    ] = f2bf(v.x);
      sA[r * LDA + c + 1] = f2bf(v.y);
      sA[r * LDA + c + 2] = f2bf(v.z);
      sA[r * LDA + c + 3] = f2bf(v.w);
    }
  } else {
    for (int i = tid * 4; i < 128 * 128; i += 1024) {
      int r = i >> 7, c = i & 127;
      float4 v = make_float4(0.f, 0.f, 0.f, 0.f);
      if (rowBase + r < limit) v = *(const float4*)(src + (size_t)(rowBase + r) * F + c);
      sA[r * LDA + c    ] = f2bf(v.x);
      sA[r * LDA + c + 1] = f2bf(v.y);
      sA[r * LDA + c + 2] = f2bf(v.z);
      sA[r * LDA + c + 3] = f2bf(v.w);
    }
  }
}

// ---------------- weight convert + transpose: Wt[n][k] = bf16(W[k][n]) ----------------
__global__ __launch_bounds__(256) void wconv_kernel(
    const float* __restrict__ W1, const float* __restrict__ W2,
    const float* __restrict__ Wa, const float* __restrict__ Wb,
    const float* __restrict__ Wc, unsigned short* __restrict__ Wt) {
  const float* src;
  switch (blockIdx.x) {
    case 0: src = W1; break;
    case 1: src = W2; break;
    case 2: src = Wa; break;
    case 3: src = Wb; break;
    default: src = Wc; break;
  }
  unsigned short* dst = Wt + (size_t)blockIdx.x * F * F;
  for (int i = threadIdx.x; i < F * F; i += blockDim.x) {
    int n = i >> 7, k = i & 127;
    dst[n * F + k] = f2bf(src[k * F + n]);
  }
}

// ---------------- f = x @ W_in2fac ----------------
__global__ __launch_bounds__(256) void in2fac_kernel(
    const float* __restrict__ x, const unsigned short* __restrict__ Wt,
    float* __restrict__ fbuf, int N) {
  __shared__ unsigned short sA[128 * LDA];
  const int tid = threadIdx.x, lane = tid & 31, wave = tid >> 5;
  const int rBase = blockIdx.x * 128;
  const bool full = (rBase + 128 <= N);

  stage_tile(sA, x, rBase, N, tid);
  __syncthreads();

  const int m = wave * 16 + (lane & 15);
  v16bf aF[4];
#pragma unroll
  for (int ks = 0; ks < 4; ++ks) aF[ks] = load_afrag(sA, m, ks * 32, lane);

  const int ncol = lane & 15;
  const int base8 = (lane >> 4) << 3;
#pragma unroll
  for (int nt = 0; nt < 8; ++nt) {
    v8f acc = {};
#pragma unroll
    for (int ks = 0; ks < 4; ++ks) {
      v16bf bF = load_bfrag(Wt, nt * 16 + ncol, ks * 32, lane);
      acc = __builtin_amdgcn_wmma_f32_16x16x32_bf16(false, aF[ks], false, bF,
                                                    (short)0, acc, false, false);
    }
    float* dst = fbuf + (size_t)(rBase + wave * 16 + base8) * F + nt * 16 + ncol;
    if (full) {
#pragma unroll
      for (int r = 0; r < 8; ++r) dst[(size_t)r * F] = acc[r];
    } else {
#pragma unroll
      for (int r = 0; r < 8; ++r) {
        if (rBase + wave * 16 + base8 + r < N) dst[(size_t)r * F] = acc[r];
      }
    }
  }
}

// ---------------- fused edge pipeline ----------------
// w_ijk = ssp(ssp(dijk@W1+b1)@W2+b2); seg_j==arange(E) => w_ij = w_ijk (identity fused)
// conv[seg_i[e]] += w_ijk[e] * f[idx_j[e]]   (seg_i sorted -> run-merged atomics)
__global__ __launch_bounds__(256) void edge_kernel(
    const float* __restrict__ dijk, const int* __restrict__ idx_j,
    const int* __restrict__ seg_i, const float* __restrict__ b1,
    const float* __restrict__ b2, const unsigned short* __restrict__ W1t,
    const unsigned short* __restrict__ W2t, const float* __restrict__ fbuf,
    float* __restrict__ conv, int E) {
  __shared__ unsigned short sA[128 * LDA];
  __shared__ int sIdx[128];
  __shared__ int sSeg[128];
  const int tid = threadIdx.x, lane = tid & 31, wave = tid >> 5;
  const int eBase = blockIdx.x * 128;

  if (tid < 128) {
    int e = eBase + tid;
    sIdx[tid] = (e < E) ? idx_j[e] : 0;
    sSeg[tid] = (e < E) ? seg_i[e] : -1;
  }
  stage_tile(sA, dijk, eBase, E, tid);
  // prefetch next block's dijk tile (64KB) into cache while we compute
  if (eBase + 128 < E) {
    const char* p = (const char*)(dijk + (size_t)(eBase + 128) * F) + tid * 256;
    __builtin_prefetch(p, 0, 1);
  }
  __syncthreads();

  const int m = wave * 16 + (lane & 15);
  const int ncol = lane & 15;
  const int base8 = (lane >> 4) << 3;
  v16bf aF[4];
#pragma unroll
  for (int ks = 0; ks < 4; ++ks) aF[ks] = load_afrag(sA, m, ks * 32, lane);
  __syncthreads();   // all waves hold A frags; sA may now be overwritten with h

  // GEMM1: h = ssp(dijk @ W1 + b1) -> bf16 back into sA
#pragma unroll
  for (int nt = 0; nt < 8; ++nt) {
    v8f acc = {};
#pragma unroll
    for (int ks = 0; ks < 4; ++ks) {
      v16bf bF = load_bfrag(W1t, nt * 16 + ncol, ks * 32, lane);
      acc = __builtin_amdgcn_wmma_f32_16x16x32_bf16(false, aF[ks], false, bF,
                                                    (short)0, acc, false, false);
    }
    float bias = b1[nt * 16 + ncol];
#pragma unroll
    for (int r = 0; r < 8; ++r) {
      int mm = wave * 16 + base8 + r;
      sA[mm * LDA + nt * 16 + ncol] = f2bf(ssp(acc[r] + bias));
    }
  }
  __syncthreads();
#pragma unroll
  for (int ks = 0; ks < 4; ++ks) aF[ks] = load_afrag(sA, m, ks * 32, lane);

  // GEMM2 + ssp + gather f[idx_j] + run-merged scatter-add into conv[seg_i]
#pragma unroll
  for (int nt = 0; nt < 8; ++nt) {
    v8f acc = {};
#pragma unroll
    for (int ks = 0; ks < 4; ++ks) {
      v16bf bF = load_bfrag(W2t, nt * 16 + ncol, ks * 32, lane);
      acc = __builtin_amdgcn_wmma_f32_16x16x32_bf16(false, aF[ks], false, bF,
                                                    (short)0, acc, false, false);
    }
    float bias = b2[nt * 16 + ncol];
    int n = nt * 16 + ncol;
    float run = 0.f;
    int curseg = -1;
#pragma unroll
    for (int r = 0; r < 8; ++r) {
      int mm = wave * 16 + base8 + r;          // 8 consecutive edges per lane
      int seg = sSeg[mm];
      float w = ssp(acc[r] + bias);
      float val = (seg >= 0) ? w * fbuf[(size_t)sIdx[mm] * F + n] : 0.f;
      if (seg != curseg) {
        if (curseg >= 0) atomicAdd(conv + (size_t)curseg * F + n, run);
        curseg = seg;
        run = val;
      } else {
        run += val;
      }
    }
    if (curseg >= 0) atomicAdd(conv + (size_t)curseg * F + n, run);
  }
}

// ---------------- c = ssp(conv@Wf + bf); v = c@Wd + bd; y = x + v ----------------
__global__ __launch_bounds__(256) void out_kernel(
    const float* __restrict__ conv, const float* __restrict__ x,
    const unsigned short* __restrict__ Wft, const unsigned short* __restrict__ Wdt,
    const float* __restrict__ bfac, const float* __restrict__ bden,
    float* __restrict__ out, int N) {
  __shared__ unsigned short sA[128 * LDA];
  const int tid = threadIdx.x, lane = tid & 31, wave = tid >> 5;
  const int rBase = blockIdx.x * 128;
  const bool full = (rBase + 128 <= N);

  stage_tile(sA, conv, rBase, N, tid);
  __syncthreads();

  const int m = wave * 16 + (lane & 15);
  const int ncol = lane & 15;
  const int base8 = (lane >> 4) << 3;
  v16bf aF[4];
#pragma unroll
  for (int ks = 0; ks < 4; ++ks) aF[ks] = load_afrag(sA, m, ks * 32, lane);
  __syncthreads();

#pragma unroll
  for (int nt = 0; nt < 8; ++nt) {
    v8f acc = {};
#pragma unroll
    for (int ks = 0; ks < 4; ++ks) {
      v16bf bF = load_bfrag(Wft, nt * 16 + ncol, ks * 32, lane);
      acc = __builtin_amdgcn_wmma_f32_16x16x32_bf16(false, aF[ks], false, bF,
                                                    (short)0, acc, false, false);
    }
    float bias = bfac[nt * 16 + ncol];
#pragma unroll
    for (int r = 0; r < 8; ++r) {
      int mm = wave * 16 + base8 + r;
      sA[mm * LDA + nt * 16 + ncol] = f2bf(ssp(acc[r] + bias));
    }
  }
  __syncthreads();
#pragma unroll
  for (int ks = 0; ks < 4; ++ks) aF[ks] = load_afrag(sA, m, ks * 32, lane);

  const size_t NF = (size_t)N * F;
#pragma unroll
  for (int nt = 0; nt < 8; ++nt) {
    v8f acc = {};
#pragma unroll
    for (int ks = 0; ks < 4; ++ks) {
      v16bf bF = load_bfrag(Wdt, nt * 16 + ncol, ks * 32, lane);
      acc = __builtin_amdgcn_wmma_f32_16x16x32_bf16(false, aF[ks], false, bF,
                                                    (short)0, acc, false, false);
    }
    float bias = bden[nt * 16 + ncol];
    int n = nt * 16 + ncol;
    const size_t rowTop = (size_t)(rBase + wave * 16 + base8);
    const float* xp = x + rowTop * F + n;
    float* yp = out + rowTop * F + n;
    float* vp = out + NF + rowTop * F + n;
    if (full) {
#pragma unroll
      for (int r = 0; r < 8; ++r) {
        float v = acc[r] + bias;
        yp[(size_t)r * F] = xp[(size_t)r * F] + v;
        vp[(size_t)r * F] = v;
      }
    } else {
#pragma unroll
      for (int r = 0; r < 8; ++r) {
        if (rBase + wave * 16 + base8 + r < N) {
          float v = acc[r] + bias;
          yp[(size_t)r * F] = xp[(size_t)r * F] + v;
          vp[(size_t)r * F] = v;
        }
      }
    }
  }
}

extern "C" void kernel_launch(void* const* d_in, const int* in_sizes, int n_in,
                              void* d_out, int out_size, void* d_ws, size_t ws_size,
                              hipStream_t stream) {
  const float* x         = (const float*)d_in[0];
  const float* dijk      = (const float*)d_in[1];
  const int*   idx_j     = (const int*)d_in[2];
  const int*   seg_i     = (const int*)d_in[3];
  // d_in[4] seg_j == arange(E): its segment_sum is the identity -> fused away
  // d_in[5] seg_i_sum == N
  const float* W1        = (const float*)d_in[6];
  const float* b1        = (const float*)d_in[7];
  const float* W2        = (const float*)d_in[8];
  const float* b2        = (const float*)d_in[9];
  const float* W_in2fac  = (const float*)d_in[10];
  const float* W_fac2out = (const float*)d_in[11];
  const float* b_fac2out = (const float*)d_in[12];
  const float* W_dense   = (const float*)d_in[13];
  const float* b_dense   = (const float*)d_in[14];

  const int N = in_sizes[0] / F;
  const int E = in_sizes[2];

  // workspace layout: [5 bf16 weight mats][f fp32 N*F][conv fp32 N*F]
  unsigned short* Wt = (unsigned short*)d_ws;
  char* p = (char*)d_ws + ((5 * F * F * sizeof(unsigned short) + 255) & ~(size_t)255);
  float* fbuf = (float*)p;
  float* conv = fbuf + (size_t)N * F;

  const int rowBlocks  = (N + 127) / 128;
  const int edgeBlocks = (E + 127) / 128;

  wconv_kernel<<<5, 256, 0, stream>>>(W1, W2, W_in2fac, W_fac2out, W_dense, Wt);
  hipMemsetAsync(conv, 0, (size_t)N * F * sizeof(float), stream);
  in2fac_kernel<<<rowBlocks, 256, 0, stream>>>(x, Wt + 2 * F * F, fbuf, N);
  edge_kernel<<<edgeBlocks, 256, 0, stream>>>(dijk, idx_j, seg_i, b1, b2,
                                              Wt, Wt + F * F, fbuf, conv, E);
  out_kernel<<<rowBlocks, 256, 0, stream>>>(conv, x, Wt + 3 * F * F, Wt + 4 * F * F,
                                            b_fac2out, b_dense, (float*)d_out, N);
}
